// RNN_43044162240701
// MI455X (gfx1250) — compile-verified, matrix operand order checked by reading, not run on hardware
//
#include <hip/hip_runtime.h>

// ---- CDNA5 WMMA types -------------------------------------------------------
typedef __attribute__((ext_vector_type(16))) __bf16        v16bf;
typedef __attribute__((ext_vector_type(8)))  float         v8f;
typedef __attribute__((ext_vector_type(4))) unsigned int   u32x4;
typedef int v4i __attribute__((vector_size(16)));           // matches async builtin param

union Frag {
    v16bf v;
    u32x4 q[2];
};

#if defined(__has_builtin)
#if __has_builtin(__builtin_amdgcn_global_load_async_to_lds_b128)
#define HAVE_ASYNC_LDS 1
#endif
#if __has_builtin(__builtin_amdgcn_s_wait_asynccnt)
#define HAVE_WAIT_ASYNC 1
#endif
#endif

// 16-byte global -> LDS copy; async (no VGPR round-trip, ASYNCcnt) when available
static __device__ __forceinline__ void copy16_g2l(const unsigned short* g,
                                                  unsigned short* l) {
#ifdef HAVE_ASYNC_LDS
    __builtin_amdgcn_global_load_async_to_lds_b128(
        (__attribute__((address_space(1))) v4i*)g,
        (__attribute__((address_space(3))) v4i*)l, 0, 0);
#else
    *(u32x4*)l = *(const u32x4*)g;
#endif
}

static __device__ __forceinline__ void wait_g2l() {
#ifdef HAVE_ASYNC_LDS
#ifdef HAVE_WAIT_ASYNC
    __builtin_amdgcn_s_wait_asynccnt(0);
#else
    asm volatile("s_wait_asynccnt 0x0" ::: "memory");
#endif
#endif
}

static __device__ __forceinline__ unsigned short f2bf(float f) {
    unsigned int u = __float_as_uint(f);
    u += 0x7FFFu + ((u >> 16) & 1u);   // round-to-nearest-even
    return (unsigned short)(u >> 16);
}

#define Bsz   8
#define Ssz   4096
#define DIN   1024
#define DHID  1024
#define DOUT  1024
#define DCAT  1040
#define KPAD  1056          // 1040 padded to multiple of 32
#define MROWS (Bsz * Ssz)   // 32768

// ---- prep: xc = [x | PE16 | pad] as bf16 -----------------------------------
__global__ __launch_bounds__(256) void prep_xc(const float* __restrict__ x,
                                               unsigned short* __restrict__ xc) {
    const int m = blockIdx.x;            // token row, 0..32767
    const int s = m & (Ssz - 1);         // position in sequence
    const float* xr = x + (size_t)m * DIN;
    unsigned short* row = xc + (size_t)m * KPAD;
    const float kfac = -logf(10000.0f) / 16.0f;
    for (int c = threadIdx.x; c < KPAD; c += 256) {
        float v;
        if (c < DIN) {
            v = xr[c];
        } else if (c < DCAT) {
            int p    = c - DIN;          // 0..15
            int pair = p >> 1;
            float dv  = expf((float)(2 * pair) * kfac);
            float ang = (float)s * dv;
            v = (p & 1) ? cosf(ang) : sinf(ang);
        } else {
            v = 0.0f;                    // K padding
        }
        row[c] = f2bf(v);
    }
}

// ---- prep: fused weights [W_dx | W_in] (KPADx2048), W_out bf16, fused bias --
__global__ __launch_bounds__(256) void prep_w(const float* __restrict__ Wdx,
                                              const float* __restrict__ bdx,
                                              const float* __restrict__ Win,
                                              const float* __restrict__ bin,
                                              const float* __restrict__ Wout,
                                              unsigned short* __restrict__ Wcat,
                                              unsigned short* __restrict__ Woutb,
                                              float* __restrict__ bcat) {
    const int blk = blockIdx.x;
    if (blk < KPAD) {                                    // Wcat row
        const int k = blk;
        unsigned short* row = Wcat + (size_t)k * 2048;
        for (int c = threadIdx.x; c < 2048; c += 256) {
            float v = 0.0f;
            if (k < DCAT)
                v = (c < DOUT) ? Wdx[(size_t)k * DOUT + c]
                               : Win[(size_t)k * DHID + (c - DOUT)];
            row[c] = f2bf(v);
        }
    } else if (blk < KPAD + DHID) {                      // Wout row
        const int k = blk - KPAD;
        unsigned short* row = Woutb + (size_t)k * DOUT;
        for (int c = threadIdx.x; c < DOUT; c += 256)
            row[c] = f2bf(Wout[(size_t)k * DOUT + c]);
    } else {                                             // fused bias
        for (int c = threadIdx.x; c < 2048; c += 256)
            bcat[c] = (c < DOUT) ? bdx[c] : bin[c - DOUT];
    }
}

// ---- tiled bf16 WMMA GEMM: C = scale*(A@B + bias + add) ---------------------
// 128x128 block tile, 256 threads = 8 wave32, each wave 32x64 (2x4 WMMA tiles)
// Double-buffered LDS, async global->LDS for A, one barrier per K-step.
__global__ __launch_bounds__(256) void gemm_bf16_wmma(
        const unsigned short* __restrict__ A,   // MxK row-major bf16
        const unsigned short* __restrict__ B,   // KxN row-major bf16
        int K, int lda, int ldb,
        const float* __restrict__ bias,         // len N
        const float* __restrict__ addsrc,       // Mxldadd f32 or nullptr
        int ldadd, float scale,
        float* __restrict__ C, int ldc) {
    __shared__ unsigned short As[2][128][32];   // [buf][m][k]
    __shared__ unsigned short Bs[2][128][32];   // [buf][n][k] (transposed)

    const int t      = threadIdx.x;
    const int wave   = t >> 5;
    const int lane   = t & 31;
    const int lane15 = lane & 15;
    const int lanehi = lane >> 4;
    const int wm     = wave & 3;                // 4 waves along M
    const int wn     = wave >> 2;               // 2 waves along N
    const int blockM = blockIdx.y * 128;
    const int blockN = blockIdx.x * 128;

    // staging coordinates (fixed per thread)
    const int am  = t >> 1;                     // A row within tile
    const int ako = (t & 1) << 4;               // A k-offset (16 elems = 32B)

    auto stage = [&](int kk, int buf) {
        const int k0 = kk << 5;
        // A tile (128x32): 2 x 16B per thread, layout-preserving -> async path
        const unsigned short* ga = A + (size_t)(blockM + am) * lda + k0 + ako;
        copy16_g2l(ga,     &As[buf][am][ako]);
        copy16_g2l(ga + 8, &As[buf][am][ako + 8]);
        // B tile (32x128) -> transposed Bs[n][k]; b128 global reads, scalar scatter
        #pragma unroll
        for (int gidx = 0; gidx < 2; ++gidx) {
            const int g  = t + gidx * 256;      // 0..511 groups of 8 elems
            const int k  = g >> 4;              // 0..31
            const int n8 = (g & 15) << 3;       // 0,8,...,120
            const u32x4 vv = *(const u32x4*)(B + (size_t)(k0 + k) * ldb + blockN + n8);
            const unsigned short* sv = (const unsigned short*)&vv;
            #pragma unroll
            for (int i = 0; i < 8; ++i)
                Bs[buf][n8 + i][k] = sv[i];
        }
    };

    v8f acc[2][4] = {};
    const int nk = K >> 5;

    stage(0, 0);
    for (int kk = 0; kk < nk; ++kk) {
        const int buf = kk & 1;
        wait_g2l();                 // own async A-chunks for `buf` complete
        __syncthreads();            // everyone's LDS writes visible
        if (kk + 1 < nk) stage(kk + 1, buf ^ 1);   // prefetch next tile

        // ---- fragments per ISA 16-bit A layout:
        // lane element j<8  : K = lanehi*8 + j
        // lane element j>=8 : K = 16 + lanehi*8 + (j-8)
        Frag af[2], bfr[4];
        #pragma unroll
        for (int i = 0; i < 2; ++i) {
            const u32x4* p = (const u32x4*)(&As[buf][wm * 32 + i * 16 + lane15][0]);
            af[i].q[0] = p[lanehi];
            af[i].q[1] = p[2 + lanehi];
        }
        #pragma unroll
        for (int j = 0; j < 4; ++j) {
            const u32x4* p = (const u32x4*)(&Bs[buf][wn * 64 + j * 16 + lane15][0]);
            bfr[j].q[0] = p[lanehi];
            bfr[j].q[1] = p[2 + lanehi];
        }
        #pragma unroll
        for (int i = 0; i < 2; ++i)
            #pragma unroll
            for (int j = 0; j < 4; ++j)
                acc[i][j] = __builtin_amdgcn_wmma_f32_16x16x32_bf16(
                    false, af[i].v, false, bfr[j].v,
                    (short)0, acc[i][j], false, false);
    }

    // ---- epilogue: C/D layout: vgpr r -> M = r + lanehi*8, N = lane15
    #pragma unroll
    for (int i = 0; i < 2; ++i) {
        #pragma unroll
        for (int j = 0; j < 4; ++j) {
            const int gn = blockN + wn * 64 + j * 16 + lane15;
            const float bv = bias[gn];
            #pragma unroll
            for (int r = 0; r < 8; ++r) {
                const int gm = blockM + wm * 32 + i * 16 + lanehi * 8 + r;
                float v = acc[i][j][r] + bv;
                if (addsrc) v += addsrc[(size_t)gm * ldadd + gn];
                C[(size_t)gm * ldc + gn] = v * scale;
            }
        }
    }
}

// ---- linear recurrence: h_t = a*h_{t-1} + sqrt(1-a^2)*u_t -------------------
// one thread per (batch, hidden-dim); reads u (f32, col 1024.. of C1), emits bf16 h
__global__ __launch_bounds__(256) void scan_kernel(const float* __restrict__ a_logit,
                                                   const float* __restrict__ C1,
                                                   unsigned short* __restrict__ hbf) {
    const int g = blockIdx.x * 256 + threadIdx.x;   // 0..8191
    const int d = g & (DHID - 1);
    const int b = g >> 10;
    const float al  = a_logit[d];
    const float a   = 1.0f / (1.0f + expf(-al));
    const float gsc = sqrtf(fmaxf(0.0f, 1.0f - a * a));
    const float* up = C1 + (size_t)b * Ssz * 2048 + DOUT + d;
    unsigned short* hp = hbf + (size_t)b * Ssz * DHID + d;
    float h = 0.0f;
    for (int s = 0; s < Ssz; ++s) {
        if (s + 16 < Ssz)
            __builtin_prefetch((const void*)(up + (size_t)(s + 16) * 2048), 0, 0);
        const float u = up[(size_t)s * 2048];
        h = fmaf(a, h, gsc * u);
        hp[(size_t)s * DHID] = f2bf(h);
    }
}

// ---- host side --------------------------------------------------------------
extern "C" void kernel_launch(void* const* d_in, const int* in_sizes, int n_in,
                              void* d_out, int out_size, void* d_ws, size_t ws_size,
                              hipStream_t stream) {
    (void)in_sizes; (void)n_in; (void)out_size; (void)ws_size;
    const float* x       = (const float*)d_in[0];
    const float* a_logit = (const float*)d_in[1];
    const float* W_dx    = (const float*)d_in[2];
    const float* b_dx    = (const float*)d_in[3];
    const float* W_in    = (const float*)d_in[4];
    const float* b_in    = (const float*)d_in[5];
    const float* W_out   = (const float*)d_in[6];
    const float* b_out   = (const float*)d_in[7];
    float* out = (float*)d_out;

    char* ws = (char*)d_ws;
    size_t off = 0;
    auto alloc = [&](size_t bytes) -> void* {
        void* p = ws + off;
        off += (bytes + 255) & ~(size_t)255;
        return p;
    };
    unsigned short* xc    = (unsigned short*)alloc((size_t)MROWS * KPAD * 2);
    unsigned short* Wcat  = (unsigned short*)alloc((size_t)KPAD * 2048 * 2);
    unsigned short* Woutb = (unsigned short*)alloc((size_t)DHID * DOUT * 2);
    float*          bcat  = (float*)alloc(2048 * 4);
    float*          C1    = (float*)alloc((size_t)MROWS * 2048 * 4);  // [dx | u]
    unsigned short* hbf   = (unsigned short*)alloc((size_t)MROWS * DHID * 2);

    // 1) build bf16 inputs/weights
    prep_xc<<<MROWS, 256, 0, stream>>>(x, xc);
    prep_w<<<KPAD + DHID + 1, 256, 0, stream>>>(W_dx, b_dx, W_in, b_in, W_out,
                                                Wcat, Woutb, bcat);

    // 2) fused GEMM1+2: [dx | u_raw] = xc @ [W_dx|W_in] + [b_dx|b_in]
    {
        dim3 grid(2048 / 128, MROWS / 128);
        gemm_bf16_wmma<<<grid, 256, 0, stream>>>(xc, Wcat, KPAD, KPAD, 2048,
                                                 bcat, nullptr, 0, 1.0f,
                                                 C1, 2048);
    }

    // 3) recurrence over S (f32), emit bf16 h
    scan_kernel<<<(Bsz * DHID) / 256, 256, 0, stream>>>(a_logit, C1, hbf);

    // 4) GEMM3 + fused epilogue: out = 0.5*(h @ W_out + b_out + dx)
    {
        dim3 grid(DOUT / 128, MROWS / 128);
        gemm_bf16_wmma<<<grid, 256, 0, stream>>>(hbf, Woutb, DHID, DHID, DOUT,
                                                 b_out, C1, 2048, 0.5f,
                                                 out, DOUT);
    }
}
